// RNNModel_91216515432948
// MI455X (gfx1250) — matrix operand order repaired and checked by reading, not compiled, not run on hardware
//
#include <hip/hip_runtime.h>

typedef __attribute__((ext_vector_type(16))) __bf16 v16bf;
typedef __attribute__((ext_vector_type(8)))  float  v8f;

#define T_STEPS 128
#define BATCH   64
#define VOCAB   10000
#define VPAD    10240
#define NINP_SZ 1024
#define ASTR    40   // LDS A-tile row stride (elements); 80B rows -> every 16B slot aligned

// ---- float -> bf16 (round to nearest even) ----
__device__ __forceinline__ __bf16 f2bf(float f) {
  union { float f; unsigned u; } in; in.f = f;
  unsigned u = in.u;
  unsigned r = (u + 0x7FFFu + ((u >> 16) & 1u)) >> 16;
  union { unsigned short s; __bf16 b; } out; out.s = (unsigned short)r;
  return out.b;
}

// ---- async 16B copy global -> LDS (CDNA5, tracked by ASYNCcnt) ----
__device__ __forceinline__ void async_copy_b128(unsigned lds_byte_off, const void* gsrc) {
  asm volatile("global_load_async_to_lds_b128 %0, %1, off"
               :: "v"(lds_byte_off), "v"(gsrc)
               : "memory");
}

// =====================================================================
// Pack fp32 weight [K, N] row-major into WMMA B-fragment tiles (bf16).
// Tile = 32(K) x 16(N).  This code's contiguous K order: lane 0-15 -> N=lane, K=0..15 (elems 0..15);
// lane 16-31 -> N=lane-16, K=16..31.  One tile = 32 lanes * 16 bf16 = 512 elems.
// Tiles ordered tile = kc * (Npad/16) + nt.  N padded with zeros to Npad.
// =====================================================================
__global__ __launch_bounds__(256) void pack_weight(
    const float* __restrict__ W, __bf16* __restrict__ out,
    int K, int N, int Npad)
{
  const int lane = threadIdx.x & 31;
  const int tile = blockIdx.x * 8 + (threadIdx.x >> 5);
  const int ntilesN = Npad >> 4;
  const int ktiles  = K >> 5;
  if (tile >= ktiles * ntilesN) return;
  const int kc = tile / ntilesN;
  const int nt = tile - kc * ntilesN;
  const int n  = nt * 16 + (lane & 15);
  const int kb = kc * 32 + (lane >> 4) * 16;
  v16bf v;
#pragma unroll
  for (int e = 0; e < 16; ++e) {
    float x = (n < N) ? W[(size_t)(kb + e) * N + n] : 0.0f;
    v[e] = f2bf(x);
  }
  *(v16bf*)(out + (size_t)tile * 512 + lane * 16) = v;
}

// ---- embedding gather fp32 -> bf16 ----
__global__ void embed_k(const int* __restrict__ tok, const float* __restrict__ encW,
                        __bf16* __restrict__ emb)
{
  const int row = blockIdx.x;        // t*B + b
  const int t = tok[row];
  const float* src = encW + (size_t)t * NINP_SZ;
  __bf16* dst = emb + (size_t)row * NINP_SZ;
  for (int e = threadIdx.x; e < NINP_SZ; e += 256) dst[e] = f2bf(src[e]);
}

// ---- copy initial h (->bf16) and c (fp32) into workspace state ----
__global__ void init_state_k(const float* __restrict__ h, const float* __restrict__ cin,
                             __bf16* __restrict__ hb, float* __restrict__ cs, int n)
{
  int i = blockIdx.x * 256 + threadIdx.x;
  if (i < n) { hb[i] = f2bf(h[i]); cs[i] = cin[i]; }
}

// =====================================================================
// GEMM: C[M, Nout](fp32) = A1[M,K1](bf16) @ B1 + A2[M,K2](bf16) @ B2 + bias
// B1/B2 pre-packed in fragment tile layout (Npad columns). A2 may be null.
// Mtotal must be a multiple of 64. Block = 256 threads = 8 waves computing
// a 64(M) x 128(N) tile: wave w -> N-subtile w, 4 accumulators over M.
// A chunks (64x32) staged with double-buffered GLOBAL_LOAD_ASYNC_TO_LDS_B128
// so the next chunk's copy overlaps the current chunk's WMMAs.
// =====================================================================
__global__ __launch_bounds__(256) void gemm_bf16_wmma(
    const __bf16* __restrict__ A1, int lda1, int K1, const __bf16* __restrict__ B1,
    const __bf16* __restrict__ A2, int lda2, int K2, const __bf16* __restrict__ B2,
    const float* __restrict__ bias, float* __restrict__ C,
    int ldc, int Nout, int Npad)
{
  __shared__ __bf16 lds_a[2][64 * ASTR];
  const int lane = threadIdx.x & 31;
  const int wave = threadIdx.x >> 5;
  const int half = lane >> 4;
  const int nr   = lane & 15;
  const int ntilesN = Npad >> 4;
  const int ntile = blockIdx.x * 8 + wave;
  const int m0 = blockIdx.y * 64;

  const int rr = threadIdx.x >> 2;        // staging: row 0..63
  const int cc = (threadIdx.x & 3) << 3;  // staging: col 0,8,16,24

  const int nc1 = K1 >> 5;
  const int nc2 = A2 ? (K2 >> 5) : 0;
  const int nc  = nc1 + nc2;

  const __bf16* a1row = A1 + (size_t)(m0 + rr) * lda1 + cc;
  const __bf16* a2row = A2 ? (A2 + (size_t)(m0 + rr) * lda2 + cc) : (const __bf16*)0;

  auto gsrc = [&](int i) -> const __bf16* {
    return (i < nc1) ? (a1row + (i << 5)) : (a2row + ((i - nc1) << 5));
  };
  auto bsrc = [&](int i) -> const __bf16* {
    return (i < nc1)
      ? (B1 + ((size_t)i * ntilesN + ntile) * 512 + lane * 16)
      : (B2 + ((size_t)(i - nc1) * ntilesN + ntile) * 512 + lane * 16);
  };

  v8f acc[4] = {};

  unsigned ldso[2];
  ldso[0] = (unsigned)(size_t)&lds_a[0][rr * ASTR + cc];
  ldso[1] = (unsigned)(size_t)&lds_a[1][rr * ASTR + cc];

  // prologue: stage chunk 0 into buffer 0
  async_copy_b128(ldso[0], gsrc(0));

  for (int i = 0; i < nc; ++i) {
    const int cur = i & 1;
    if (i + 1 < nc) {
      // prefetch next chunk into the other buffer, then wait for current only
      async_copy_b128(ldso[cur ^ 1], gsrc(i + 1));
      asm volatile("s_wait_asynccnt 0x1" ::: "memory");
      __builtin_prefetch(bsrc(i + 1), 0, 0);      // global_prefetch next B tile
    } else {
      asm volatile("s_wait_asynccnt 0x0" ::: "memory");
    }
    __syncthreads();

    // B fragment: one contiguous 32-byte load per lane from packed tiles
    const v16bf bfrag = *(const v16bf*)bsrc(i);

    const __bf16* abase = &lds_a[cur][0];
#pragma unroll
    for (int mt = 0; mt < 4; ++mt) {
      const __bf16* ap = abase + (mt * 16 + nr) * ASTR;
      v16bf afrag;
#pragma unroll
      for (int v = 0; v < 8; ++v) {
        // A layout matching packed B: VGPR v -> K=2v(+16*half)
        const int kb = 2 * v + half * 16;
        afrag[2 * v]     = ap[kb];
        afrag[2 * v + 1] = ap[kb + 1];
      }
      acc[mt] = __builtin_amdgcn_wmma_f32_16x16x32_bf16(
          false, afrag, false, bfrag, (short)0, acc[mt], false, false);
    }
    __syncthreads();   // all reads of lds_a[cur] done before it is re-targeted
  }

  const int ncol = ntile * 16 + nr;
  if (ncol < Nout) {
    const float bv = bias ? bias[ncol] : 0.0f;
#pragma unroll
    for (int mt = 0; mt < 4; ++mt) {
#pragma unroll
      for (int r = 0; r < 8; ++r) {
        // ISA C/D layout: VGPR r -> M = r + 8*half, N = lane&15
        const int m = m0 + mt * 16 + half * 8 + r;
        C[(size_t)m * ldc + ncol] = acc[mt][r] + bv;
      }
    }
  }
}

// ---- block-wide helpers for the cell ----
__device__ __forceinline__ float block_max(float v, float* s) {
  const int t = threadIdx.x;
  s[t] = v; __syncthreads();
  for (int o = 128; o > 0; o >>= 1) {
    if (t < o) s[t] = fmaxf(s[t], s[t + o]);
    __syncthreads();
  }
  float r = s[0];
  __syncthreads();
  return r;
}

__device__ __forceinline__ float block_scan_incl(float v, float* s, float* total) {
  const int t = threadIdx.x;
  s[t] = v; __syncthreads();
#pragma unroll
  for (int o = 1; o < 256; o <<= 1) {
    float add = (t >= o) ? s[t - o] : 0.0f;
    __syncthreads();
    s[t] += add;
    __syncthreads();
  }
  float inc = s[t];
  *total = s[255];
  __syncthreads();
  return inc;
}

// =====================================================================
// ON-LSTM cell: gates [B, 6H] fp32 -> updates c (fp32) and h (bf16).
// One block per batch row; thread handles H/256 contiguous elements.
// cumax(softmax) implemented as exp/scan/normalize (max-stabilized).
// =====================================================================
__global__ __launch_bounds__(256) void onlstm_cell_k(
    const float* __restrict__ gates, float* __restrict__ c,
    __bf16* __restrict__ hb, __bf16* __restrict__ hout, int H)
{
  __shared__ float s[256];
  const int b  = blockIdx.x;
  const int PE = H >> 8;              // 8 (H=2048) or 4 (H=1024)
  const int e0 = threadIdx.x * PE;
  const float* g = gates + (size_t)b * 6 * H;

  float cf[8], ci[8];

  {  // cforget = cumsum(softmax(fm)),  fm = g[0:H]
    float x[8]; float lmax = -1e30f;
    for (int j = 0; j < PE; ++j) { x[j] = g[e0 + j]; lmax = fmaxf(lmax, x[j]); }
    const float bmax = block_max(lmax, s);
    float run = 0.0f;
    for (int j = 0; j < PE; ++j) { run += __expf(x[j] - bmax); cf[j] = run; }
    float tot; const float inc = block_scan_incl(run, s, &tot);
    const float off = inc - run, inv = 1.0f / tot;
    for (int j = 0; j < PE; ++j) cf[j] = (off + cf[j]) * inv;
  }
  {  // cin = 1 - cumsum(softmax(im)),  im = g[H:2H]
    float x[8]; float lmax = -1e30f;
    for (int j = 0; j < PE; ++j) { x[j] = g[H + e0 + j]; lmax = fmaxf(lmax, x[j]); }
    const float bmax = block_max(lmax, s);
    float run = 0.0f;
    for (int j = 0; j < PE; ++j) { run += __expf(x[j] - bmax); ci[j] = run; }
    float tot; const float inc = block_scan_incl(run, s, &tot);
    const float off = inc - run, inv = 1.0f / tot;
    for (int j = 0; j < PE; ++j) ci[j] = 1.0f - (off + ci[j]) * inv;
  }

  for (int j = 0; j < PE; ++j) {
    const int e = e0 + j;
    const float iv = 1.0f / (1.0f + __expf(-g[2 * H + e]));
    const float fv = 1.0f / (1.0f + __expf(-g[3 * H + e]));
    const float gv = tanhf(g[4 * H + e]);
    const float ov = 1.0f / (1.0f + __expf(-g[5 * H + e]));
    const float ovl = cf[j] * ci[j];
    const float fh = fv * ovl + (cf[j] - ovl);
    const float ih = iv * ovl + (ci[j] - ovl);
    const size_t idx = (size_t)b * H + e;
    const float cn = fh * c[idx] + ih * gv;
    c[idx] = cn;
    const float hn = ov * tanhf(cn);
    const __bf16 hv = f2bf(hn);
    hb[idx] = hv;
    if (hout) hout[idx] = hv;
  }
}

// =====================================================================
extern "C" void kernel_launch(void* const* d_in, const int* in_sizes, int n_in,
                              void* d_out, int out_size, void* d_ws, size_t ws_size,
                              hipStream_t stream)
{
  (void)in_sizes; (void)n_in; (void)out_size; (void)ws_size;
  const int*   tokens = (const int*)  d_in[0];
  const float* encW   = (const float*)d_in[1];
  const float* decW   = (const float*)d_in[2];
  const float* decb   = (const float*)d_in[3];
  const float* Wih[3] = {(const float*)d_in[4],  (const float*)d_in[9],  (const float*)d_in[14]};
  const float* Whh[3] = {(const float*)d_in[5],  (const float*)d_in[10], (const float*)d_in[15]};
  const float* bgs[3] = {(const float*)d_in[6],  (const float*)d_in[11], (const float*)d_in[16]};
  const float* h0s[3] = {(const float*)d_in[7],  (const float*)d_in[12], (const float*)d_in[17]};
  const float* c0s[3] = {(const float*)d_in[8],  (const float*)d_in[13], (const float*)d_in[18]};

  const int din[3] = {1024, 2048, 2048};
  const int dh [3] = {2048, 2048, 1024};
  const int ng [3] = {6 * 2048, 6 * 2048, 6 * 1024};
  const int TB = T_STEPS * BATCH;  // 8192

  // ---- workspace carve-up (256B aligned) ----
  char* base = (char*)d_ws;
  size_t off = 0;
  auto alloc = [&](size_t bytes) -> void* {
    void* p = base + off;
    off = (off + bytes + 255) & ~(size_t)255;
    return p;
  };
  __bf16* emb  = (__bf16*)alloc((size_t)TB * NINP_SZ * 2);
  __bf16* hout = (__bf16*)alloc((size_t)TB * NINP_SZ * 2);
  __bf16 *WihP[3], *WhhP[3], *hB[3]; float* cS[3];
  for (int l = 0; l < 3; ++l) {
    WihP[l] = (__bf16*)alloc((size_t)din[l] * ng[l] * 2);
    WhhP[l] = (__bf16*)alloc((size_t)dh[l]  * ng[l] * 2);
    hB[l]   = (__bf16*)alloc((size_t)BATCH * dh[l] * 2);
    cS[l]   = (float*) alloc((size_t)BATCH * dh[l] * 4);
  }
  __bf16* decWP  = (__bf16*)alloc((size_t)NINP_SZ * VPAD * 2);
  float*  gatesB = (float*) alloc((size_t)BATCH * ng[0] * 4);

  const dim3 blk(256);

  // ---- one-time (per call) weight pack to bf16 fragment tiles ----
  for (int l = 0; l < 3; ++l) {
    int t1 = (din[l] >> 5) * (ng[l] >> 4);
    pack_weight<<<dim3((t1 + 7) / 8), blk, 0, stream>>>(Wih[l], WihP[l], din[l], ng[l], ng[l]);
    int t2 = (dh[l] >> 5) * (ng[l] >> 4);
    pack_weight<<<dim3((t2 + 7) / 8), blk, 0, stream>>>(Whh[l], WhhP[l], dh[l], ng[l], ng[l]);
  }
  {
    int t3 = (NINP_SZ >> 5) * (VPAD >> 4);
    pack_weight<<<dim3((t3 + 7) / 8), blk, 0, stream>>>(decW, decWP, NINP_SZ, VOCAB, VPAD);
  }

  // ---- embedding + state init ----
  embed_k<<<dim3(TB), blk, 0, stream>>>(tokens, encW, emb);
  for (int l = 0; l < 3; ++l) {
    int n = BATCH * dh[l];
    init_state_k<<<dim3((n + 255) / 256), blk, 0, stream>>>(h0s[l], c0s[l], hB[l], cS[l], n);
  }

  // ---- recurrence over T steps (stream order carries the dependency) ----
  for (int t = 0; t < T_STEPS; ++t) {
    const __bf16* x = emb + (size_t)t * BATCH * NINP_SZ;
    for (int l = 0; l < 3; ++l) {
      const __bf16* A1 = (l == 0) ? x : hB[l - 1];
      gemm_bf16_wmma<<<dim3(ng[l] / 128, 1), blk, 0, stream>>>(
          A1, din[l], din[l], WihP[l],
          hB[l], dh[l], dh[l], WhhP[l],
          bgs[l], gatesB, ng[l], ng[l], ng[l]);
      onlstm_cell_k<<<dim3(BATCH), blk, 0, stream>>>(
          gatesB, cS[l], hB[l],
          (l == 2) ? (hout + (size_t)t * BATCH * NINP_SZ) : (__bf16*)nullptr,
          dh[l]);
    }
  }

  // ---- decode: [8192,1024] @ [1024,10000] + b -> d_out fp32 ----
  gemm_bf16_wmma<<<dim3(VPAD / 128, TB / 64), blk, 0, stream>>>(
      hout, NINP_SZ, NINP_SZ, decWP,
      nullptr, 0, 0, nullptr,
      decb, (float*)d_out, VOCAB, VOCAB, VPAD);
}
